// MyConv2d_50955492000193
// MI455X (gfx1250) — compile-verified, matrix operand order checked by reading, not run on hardware
//
#include <hip/hip_runtime.h>

typedef __attribute__((ext_vector_type(16))) __bf16 v16bf;
typedef __attribute__((ext_vector_type(8)))  __bf16 v8bf;
typedef __attribute__((ext_vector_type(2)))  __bf16 v2bf;
typedef __attribute__((ext_vector_type(8)))  float  v8f;

#define BB   32
#define IC   128
#define IH   112
#define IW   112
#define OC   256
#define OH   56
#define OW   56
#define HW   (IH*IW)
#define OHW  (OH*OW)

#define MT   128   // M (OC) tile per block
#define NT   128   // N (pixel) tile per block
#define LDT  136   // LDS row stride in bf16 (272B: 16B-aligned, conflict-free frag reads)

// ws layout: [128 zero bf16][xp: BB*IH*IW*IC bf16][wp: 9*OC*IC bf16]
#define XP_ELEMS   ((long)BB * IH * IW * IC)          // 51,380,224
#define WP_ELEMS   (9 * OC * IC)                      // 294,912
#define ZPAD_ELEMS 128
#define WS_NEED    ((size_t)(ZPAD_ELEMS + XP_ELEMS + WP_ELEMS) * 2)

// ---------------------------------------------------------------------------
// Weight pack: f32 OIHW -> bf16 [tap][oc][k=128]; also zeroes the pad region.
// ---------------------------------------------------------------------------
__global__ __launch_bounds__(256)
void pack_weights(const float* __restrict__ w, __bf16* __restrict__ wp,
                  __bf16* __restrict__ zpad)
{
    if (blockIdx.x == 0 && threadIdx.x < ZPAD_ELEMS)
        zpad[threadIdx.x] = (__bf16)0.0f;
    const int i   = blockIdx.x * 256 + threadIdx.x;   // 0 .. WP_ELEMS-1
    const int k   = i & 127;
    const int oc  = (i >> 7) & 255;
    const int tap = i >> 15;                          // kh*3+kw
    wp[i] = (__bf16)w[((long)oc * IC + k) * 9 + tap];
}

// ---------------------------------------------------------------------------
// Activation pack: f32 NCHW -> bf16 channels-last [b][ih][iw][ic]
// ---------------------------------------------------------------------------
__global__ __launch_bounds__(256)
void pack_x(const float* __restrict__ x, __bf16* __restrict__ xp)
{
    const long o   = (long)blockIdx.x * 256 + threadIdx.x;  // pair index
    const int icp  = (int)(o & 63);
    long rest      = o >> 6;
    const int iw   = (int)(rest % IW);  rest /= IW;
    const int ih   = (int)(rest % IH);
    const int b    = (int)(rest / IH);
    const long rb  = ((long)(b * IC + 2 * icp) * IH + ih) * IW + iw;
    v2bf pk;
    pk[0] = (__bf16)x[rb];
    pk[1] = (__bf16)x[rb + HW];
    ((v2bf*)xp)[o] = pk;
}

// ---------------------------------------------------------------------------
// Main GEMM: software-pipelined over taps with LDS double-buffering.
// Per tap: prefetch next tap's A/B (8x global_load_b128 into regs) ->
// compute 4 chunk-steps x 8 WMMA from LDS[cur] -> commit regs to LDS[cur^1].
// xp0 = ws base: [0..127]=zeros, data at +128 (OOB pixels select offset 0).
// ---------------------------------------------------------------------------
__global__ __launch_bounds__(256)
void conv2d_igemm_wmma_fast(const __bf16* __restrict__ xp0,
                            const __bf16* __restrict__ wp,
                            const float* __restrict__ bias,
                            float* __restrict__ out)
{
    __shared__ __bf16 Alds[2][MT * LDT];   // 128 oc x 128 k (padded), x2 bufs
    __shared__ __bf16 Blds[2][NT * LDT];   // 128 px x 128 k (padded), x2 bufs

    const int t    = threadIdx.x;
    const int lane = t & 31;
    const int wv   = t >> 5;        // wave id 0..7
    const int wm   = wv >> 2;       // 0..1  (M sub-tile of 64)
    const int wn   = wv & 3;        // 0..3  (N sub-tile of 32)
    const int m16  = lane & 15;
    const int half = lane >> 4;

    const int N0 = blockIdx.x * NT;
    const int M0 = blockIdx.y * MT;

    // staging ownership: one row/pixel + 32 contiguous k per thread
    const int srow = t & 127;
    const int kb   = (t >> 7) * 64;   // 0 or 64

    const int ng   = N0 + srow;
    const int bimg = ng / OHW;
    const int rem  = ng % OHW;
    const int ohp  = rem / OW;
    const int owp  = rem % OW;

    v8f  acc[4][2] = {};
    v8bf pbA[4], pbB[4];              // in-flight prefetch registers

    // issue next tap's global loads into registers
    auto issue_loads = [&](int tap) {
        const int kh = tap / 3, kw = tap % 3;
        const int ih = ohp * 2 - 1 + kh;
        const int iw = owp * 2 - 1 + kw;
        const bool valid = (ih >= 0) & (ih < IH) & (iw >= 0) & (iw < IW);
        const long sel = valid
            ? (long)ZPAD_ELEMS + ((long)(bimg * IH + ih) * IW + iw) * IC : 0L;
        const __bf16* bsrc = xp0 + sel + kb;
        const __bf16* asrc = wp + ((long)tap * OC + M0 + srow) * IC + kb;
        #pragma unroll
        for (int e = 0; e < 4; ++e) pbB[e] = *(const v8bf*)(bsrc + e * 8);
        #pragma unroll
        for (int e = 0; e < 4; ++e) pbA[e] = *(const v8bf*)(asrc + e * 8);
    };
    // commit prefetch registers to LDS buffer
    auto commit = [&](int buf) {
        __bf16* bdst = &Blds[buf][srow * LDT + kb];
        __bf16* adst = &Alds[buf][srow * LDT + kb];
        #pragma unroll
        for (int e = 0; e < 4; ++e) *(v8bf*)(bdst + e * 8) = pbB[e];
        #pragma unroll
        for (int e = 0; e < 4; ++e) *(v8bf*)(adst + e * 8) = pbA[e];
    };

    issue_loads(0);
    commit(0);
    int cur = 0;

    for (int tap = 0; tap < 9; ++tap) {
        __syncthreads();              // LDS[cur] staged & visible
        if (tap < 8) issue_loads(tap + 1);   // loads fly during compute

        // ---- 4 chunk-steps from LDS[cur], 8 WMMA each ----
        #pragma unroll
        for (int chunk = 0; chunk < 4; ++chunk) {
            const int c = chunk * 32;
            v16bf afr[4], bfr[2];
            #pragma unroll
            for (int mi = 0; mi < 4; ++mi) {
                const int row = wm * 64 + mi * 16 + m16;
                v8bf lo = *(const v8bf*)&Alds[cur][row * LDT + c + 8 * half];
                v8bf hi = *(const v8bf*)&Alds[cur][row * LDT + c + 16 + 8 * half];
                afr[mi] = __builtin_shufflevector(lo, hi,
                    0,1,2,3,4,5,6,7,8,9,10,11,12,13,14,15);
            }
            #pragma unroll
            for (int ni = 0; ni < 2; ++ni) {
                const int col = wn * 32 + ni * 16 + m16;
                v8bf lo = *(const v8bf*)&Blds[cur][col * LDT + c + 8 * half];
                v8bf hi = *(const v8bf*)&Blds[cur][col * LDT + c + 16 + 8 * half];
                bfr[ni] = __builtin_shufflevector(lo, hi,
                    0,1,2,3,4,5,6,7,8,9,10,11,12,13,14,15);
            }
            #pragma unroll
            for (int mi = 0; mi < 4; ++mi)
                #pragma unroll
                for (int ni = 0; ni < 2; ++ni)
                    acc[mi][ni] = __builtin_amdgcn_wmma_f32_16x16x32_bf16(
                        false, afr[mi], false, bfr[ni],
                        (short)0, acc[mi][ni], false, false);
        }

        if (tap < 8) commit(cur ^ 1);  // stores complete before next barrier
        cur ^= 1;
    }

    // ---- epilogue: bias add + store ----
    #pragma unroll
    for (int ni = 0; ni < 2; ++ni) {
        const int nloc = wn * 32 + ni * 16 + m16;
        const int ngl  = N0 + nloc;
        const int bo   = ngl / OHW;
        const int ro   = ngl % OHW;
        float* obase = out + (long)bo * OC * OHW + ro;
        #pragma unroll
        for (int mi = 0; mi < 4; ++mi) {
            #pragma unroll
            for (int r = 0; r < 8; ++r) {
                const int m = M0 + wm * 64 + mi * 16 + r + 8 * half;
                obase[(long)m * OHW] = acc[mi][ni][r] + bias[m];
            }
        }
    }
}

// ---------------------------------------------------------------------------
// Fallback (no scratch): fully self-contained, gathers + converts per step.
// ---------------------------------------------------------------------------
#define LDA  40
#define LDB  40
__global__ __launch_bounds__(256)
void conv2d_igemm_wmma_fallback(const float* __restrict__ x,
                                const float* __restrict__ w,
                                const float* __restrict__ bias,
                                float* __restrict__ out)
{
    __shared__ __bf16 Alds[MT * LDA];
    __shared__ __bf16 Blds[NT * LDB];

    const int t    = threadIdx.x;
    const int lane = t & 31;
    const int wv   = t >> 5;
    const int wm   = wv >> 2;
    const int wn   = wv & 3;
    const int m16  = lane & 15;
    const int half = lane >> 4;

    const int N0 = blockIdx.x * NT;
    const int M0 = blockIdx.y * MT;

    const int srow = t & 127;
    const int cb   = (t >> 7) * 16;

    const int ng   = N0 + srow;
    const int bimg = ng / OHW;
    const int rem  = ng % OHW;
    const int ohp  = rem / OW;
    const int owp  = rem % OW;

    v8f acc[4][2] = {};

    for (int kh = 0; kh < 3; ++kh) {
        const int ih = ohp * 2 - 1 + kh;
        for (int kw = 0; kw < 3; ++kw) {
            const int iw = owp * 2 - 1 + kw;
            const bool valid = (ih >= 0) & (ih < IH) & (iw >= 0) & (iw < IW);
            const long xbase = valid
                ? ((long)bimg * IC * HW + (long)ih * IW + iw) : 0L;

            for (int c = 0; c < IC; c += 32) {
                __syncthreads();
                {
                    v8bf p0, p1;
                    #pragma unroll
                    for (int e = 0; e < 8; ++e) {
                        float v = x[xbase + (long)(c + cb + e) * HW];
                        p0[e] = (__bf16)(valid ? v : 0.0f);
                    }
                    #pragma unroll
                    for (int e = 0; e < 8; ++e) {
                        float v = x[xbase + (long)(c + cb + 8 + e) * HW];
                        p1[e] = (__bf16)(valid ? v : 0.0f);
                    }
                    *(v8bf*)&Blds[srow * LDB + cb]     = p0;
                    *(v8bf*)&Blds[srow * LDB + cb + 8] = p1;
                }
                {
                    const long wbase = ((long)(M0 + srow) * IC + c + cb) * 9
                                       + kh * 3 + kw;
                    v8bf p0, p1;
                    #pragma unroll
                    for (int e = 0; e < 8; ++e)
                        p0[e] = (__bf16)w[wbase + (long)e * 9];
                    #pragma unroll
                    for (int e = 0; e < 8; ++e)
                        p1[e] = (__bf16)w[wbase + (long)(8 + e) * 9];
                    *(v8bf*)&Alds[srow * LDA + cb]     = p0;
                    *(v8bf*)&Alds[srow * LDA + cb + 8] = p1;
                }
                __syncthreads();

                v16bf afr[4], bfr[2];
                #pragma unroll
                for (int mi = 0; mi < 4; ++mi) {
                    const int row = wm * 64 + mi * 16 + m16;
                    v8bf lo = *(const v8bf*)&Alds[row * LDA + 8 * half];
                    v8bf hi = *(const v8bf*)&Alds[row * LDA + 16 + 8 * half];
                    afr[mi] = __builtin_shufflevector(lo, hi,
                        0,1,2,3,4,5,6,7,8,9,10,11,12,13,14,15);
                }
                #pragma unroll
                for (int ni = 0; ni < 2; ++ni) {
                    const int col = wn * 32 + ni * 16 + m16;
                    v8bf lo = *(const v8bf*)&Blds[col * LDB + 8 * half];
                    v8bf hi = *(const v8bf*)&Blds[col * LDB + 16 + 8 * half];
                    bfr[ni] = __builtin_shufflevector(lo, hi,
                        0,1,2,3,4,5,6,7,8,9,10,11,12,13,14,15);
                }
                #pragma unroll
                for (int mi = 0; mi < 4; ++mi)
                    #pragma unroll
                    for (int ni = 0; ni < 2; ++ni)
                        acc[mi][ni] = __builtin_amdgcn_wmma_f32_16x16x32_bf16(
                            false, afr[mi], false, bfr[ni],
                            (short)0, acc[mi][ni], false, false);
            }
        }
    }

    #pragma unroll
    for (int ni = 0; ni < 2; ++ni) {
        const int nloc = wn * 32 + ni * 16 + m16;
        const int ngl  = N0 + nloc;
        const int bo   = ngl / OHW;
        const int ro   = ngl % OHW;
        float* obase = out + (long)bo * OC * OHW + ro;
        #pragma unroll
        for (int mi = 0; mi < 4; ++mi) {
            #pragma unroll
            for (int r = 0; r < 8; ++r) {
                const int m = M0 + wm * 64 + mi * 16 + r + 8 * half;
                obase[(long)m * OHW] = acc[mi][ni][r] + bias[m];
            }
        }
    }
}

extern "C" void kernel_launch(void* const* d_in, const int* in_sizes, int n_in,
                              void* d_out, int out_size, void* d_ws, size_t ws_size,
                              hipStream_t stream) {
    const float* x    = (const float*)d_in[0];
    const float* wgt  = (const float*)d_in[1];
    const float* bias = (const float*)d_in[2];
    float* outp       = (float*)d_out;

    const int NPIX = BB * OH * OW;               // 100352
    dim3 grid(NPIX / NT, OC / MT);               // (784, 2)
    dim3 block(256);

    if (ws_size >= WS_NEED) {
        __bf16* ws0 = (__bf16*)d_ws;
        __bf16* xp  = ws0 + ZPAD_ELEMS;
        __bf16* wp  = ws0 + ZPAD_ELEMS + XP_ELEMS;
        pack_weights<<<WP_ELEMS / 256, 256, 0, stream>>>(wgt, wp, ws0);
        pack_x<<<(int)(XP_ELEMS / 2 / 256), 256, 0, stream>>>(x, xp);
        conv2d_igemm_wmma_fast<<<grid, block, 0, stream>>>(ws0, wp, bias, outp);
    } else {
        conv2d_igemm_wmma_fallback<<<grid, block, 0, stream>>>(x, wgt, bias, outp);
    }
}
